// EnergyDecoder_28381143892832
// MI455X (gfx1250) — matrix operand order repaired and checked by reading, not compiled, not run on hardware
//
#include <hip/hip_runtime.h>
#include <hip/hip_bf16.h>
#include <hip/hip_fp16.h>

typedef __attribute__((ext_vector_type(16))) _Float16 v16h;
typedef __attribute__((ext_vector_type(2)))  __fp16   v2fp16;
typedef __attribute__((ext_vector_type(8)))  float    v8f;
typedef __attribute__((ext_vector_type(4)))  float    v4f;

#define NB 8
#define NL 64
#define NT 512
#define NC 128
#define K2 256   // 2*C

// d_ws layout (bytes):
//   [0, 131072)        : W1 combined, f16, pre-shuffled into WMMA B-fragment order.
//                        frag f = ntile*8 + kc (ntile 0..15, kc 0..7); halfIdx = f*512 + lane*16 + e
//                        holds W[K][Ncomb] with K = kc*32 + (lane&16) + e, Ncomb = ntile*16 + (lane&15)
//   [131072, 132096)   : b1 combined f32[256]  (bA1 | bB1)
//   [132096, 133120)   : w2 combined f32[256]  (WA2 | WB2)
//   [133120, 133248)   : accum f32[8][4]       (vdw, hbond, metal, hphob raw sums)
#define WS_B1   131072
#define WS_W2   132096
#define WS_ACC  133120

// ---------------------------------------------------------------------------
// Prep: convert+shuffle weights to f16 B-fragment order, pack b1/w2, zero accum
// ---------------------------------------------------------------------------
__global__ void prep_kernel(const float* __restrict__ WA1, const float* __restrict__ bA1,
                            const float* __restrict__ WA2, const float* __restrict__ WB1,
                            const float* __restrict__ bB1, const float* __restrict__ WB2,
                            unsigned char* __restrict__ ws) {
    _Float16* wsW  = (_Float16*)ws;
    float*    wsB1 = (float*)(ws + WS_B1);
    float*    wsW2 = (float*)(ws + WS_W2);
    float*    wsAc = (float*)(ws + WS_ACC);

    int gid = blockIdx.x * 256 + threadIdx.x;          // 65536 threads total
    int f    = gid >> 9;
    int rem  = gid & 511;
    int lane = rem >> 4;
    int e    = rem & 15;
    int ntile = f >> 3;
    int kc    = f & 7;
    int N = ntile * 16 + (lane & 15);                  // combined column 0..255
    int K = kc * 32 + (lane & 16) + e;                 // 0..255
    float v = (N < NC) ? WA1[K * NC + N] : WB1[K * NC + (N - NC)];
    wsW[gid] = (_Float16)v;

    if (gid < 256) {
        wsB1[gid] = (gid < NC) ? bA1[gid] : bB1[gid - NC];
        wsW2[gid] = (gid < NC) ? WA2[gid] : WB2[gid - NC];
    }
    if (gid < NB * 4) wsAc[gid] = 0.0f;
}

// ---------------------------------------------------------------------------
// Main kernel: one block = one (b,l) row = 512 pairs.
// 8 waves; each wave runs 2 passes of a DUAL 16-pair tile (M=32 per pass):
// one LDS B-fragment load feeds two independent WMMAs.
// ---------------------------------------------------------------------------
__launch_bounds__(256)
__global__ void energy_kernel(const float* __restrict__ h,
                              const float* __restrict__ lpos, const float* __restrict__ tpos,
                              const float* __restrict__ lrad, const float* __restrict__ trad,
                              const float* __restrict__ lnm,  const float* __restrict__ tnm,
                              const float* __restrict__ inter,
                              const float* __restrict__ bA2p, const float* __restrict__ bB2p,
                              unsigned char* __restrict__ ws) {
    extern __shared__ unsigned char smem[];
    _Float16* sW   = (_Float16*)smem;                  // 65536 halves (128 KB)
    float*    sB1  = (float*)(smem + 131072);          // 256
    float*    sW2  = sB1 + 256;                        // 256
    float*    sMLP = sW2 + 256;                        // 8 waves * 32 pairs * 2 = 512
    float*    sAcc = sMLP + 512;                       // 4

    const int tid = threadIdx.x;

    // Stage shuffled weights + b1/w2 into LDS (contiguous copy)
    {
        const uint4* src = (const uint4*)ws;
        uint4*       dst = (uint4*)smem;
        for (int i = tid; i < 131072 / 16; i += 256) dst[i] = src[i];
        const float* b1s = (const float*)(ws + WS_B1);
        for (int i = tid; i < 512; i += 256) sB1[i] = b1s[i];   // covers sB1 and sW2
        if (tid < 4) sAcc[tid] = 0.0f;
    }
    __syncthreads();

    const int w    = tid >> 5;
    const int lane = tid & 31;
    const int row  = lane & 15;                        // pair-row within 16-pair tile
    const int b    = blockIdx.x >> 6;
    const int l    = blockIdx.x & 63;

    const float bA2 = bA2p[0], bB2 = bB2p[0];
    const float lx  = lpos[(b * NL + l) * 3 + 0];
    const float ly  = lpos[(b * NL + l) * 3 + 1];
    const float lz  = lpos[(b * NL + l) * 3 + 2];
    const float rl  = lrad[b * NL + l];
    const float nml = lnm[b * NL + l];
    const size_t hbase = (size_t)(b * NL + l) * NT * K2;

    union AF { v2fp16 h2[8]; v16h v; };

    for (int tt = 0; tt < 2; ++tt) {
        const int t0 = (w * 4 + tt * 2) * 16;          // dual tile: t0..t0+15 and t0+16..t0+31
        const size_t hrow0 = hbase + (size_t)(t0 + row) * K2;
        const size_t hrow1 = hrow0 + (size_t)16 * K2;
        const int off = (lane & 16) ? 16 : 0;          // lanes 16-31 take k+16 (contiguous k order, same as W1 pack)

        if (tt == 0) {                                 // prefetch next dual-tile's rows
            __builtin_prefetch(h + hrow0 + 32 * K2, 0, 0);
            __builtin_prefetch(h + hrow1 + 32 * K2, 0, 0);
        }

        // Build A fragments for both tiles (K = 0..255 -> 8 chunks of 32), f16
        AF A0[8], A1[8];
        #pragma unroll
        for (int kc = 0; kc < 8; ++kc) {
            {
                const float* hp = h + hrow0 + kc * 32 + off;
                v4f f0 = *(const v4f*)(hp);
                v4f f1 = *(const v4f*)(hp + 4);
                v4f f2 = *(const v4f*)(hp + 8);
                v4f f3 = *(const v4f*)(hp + 12);
                A0[kc].h2[0] = __builtin_amdgcn_cvt_pkrtz(f0.x, f0.y);
                A0[kc].h2[1] = __builtin_amdgcn_cvt_pkrtz(f0.z, f0.w);
                A0[kc].h2[2] = __builtin_amdgcn_cvt_pkrtz(f1.x, f1.y);
                A0[kc].h2[3] = __builtin_amdgcn_cvt_pkrtz(f1.z, f1.w);
                A0[kc].h2[4] = __builtin_amdgcn_cvt_pkrtz(f2.x, f2.y);
                A0[kc].h2[5] = __builtin_amdgcn_cvt_pkrtz(f2.z, f2.w);
                A0[kc].h2[6] = __builtin_amdgcn_cvt_pkrtz(f3.x, f3.y);
                A0[kc].h2[7] = __builtin_amdgcn_cvt_pkrtz(f3.z, f3.w);
            }
            {
                const float* hp = h + hrow1 + kc * 32 + off;
                v4f f0 = *(const v4f*)(hp);
                v4f f1 = *(const v4f*)(hp + 4);
                v4f f2 = *(const v4f*)(hp + 8);
                v4f f3 = *(const v4f*)(hp + 12);
                A1[kc].h2[0] = __builtin_amdgcn_cvt_pkrtz(f0.x, f0.y);
                A1[kc].h2[1] = __builtin_amdgcn_cvt_pkrtz(f0.z, f0.w);
                A1[kc].h2[2] = __builtin_amdgcn_cvt_pkrtz(f1.x, f1.y);
                A1[kc].h2[3] = __builtin_amdgcn_cvt_pkrtz(f1.z, f1.w);
                A1[kc].h2[4] = __builtin_amdgcn_cvt_pkrtz(f2.x, f2.y);
                A1[kc].h2[5] = __builtin_amdgcn_cvt_pkrtz(f2.z, f2.w);
                A1[kc].h2[6] = __builtin_amdgcn_cvt_pkrtz(f3.x, f3.y);
                A1[kc].h2[7] = __builtin_amdgcn_cvt_pkrtz(f3.z, f3.w);
            }
        }

        // Layer 1 (WMMA, dual tile) fused with layer 2 (per-N dot with w2)
        float sA0[8] = {0,0,0,0,0,0,0,0}, sB0[8] = {0,0,0,0,0,0,0,0};
        float sA1[8] = {0,0,0,0,0,0,0,0}, sB1r[8] = {0,0,0,0,0,0,0,0};
        for (int ntile = 0; ntile < 16; ++ntile) {
            v8f acc0 = {}, acc1 = {};
            const _Float16* wbase = sW + (size_t)(ntile * 8) * 512 + lane * 16;
            #pragma unroll
            for (int kc = 0; kc < 8; ++kc) {
                v16h bf = *(const v16h*)(wbase + kc * 512);
                acc0 = __builtin_amdgcn_wmma_f32_16x16x32_f16(
                           false, A0[kc].v, false, bf, (short)0, acc0, false, false);
                acc1 = __builtin_amdgcn_wmma_f32_16x16x32_f16(
                           false, A1[kc].v, false, bf, (short)0, acc1, false, false);
            }
            const int Ncomb = ntile * 16 + row;        // D: N = lane%16, M = i + 8*(lane>=16)
            const float b1 = sB1[Ncomb], w2 = sW2[Ncomb];
            float* d0 = (ntile < 8) ? sA0 : sB0;
            float* d1 = (ntile < 8) ? sA1 : sB1r;
            #pragma unroll
            for (int i = 0; i < 8; ++i) {
                float r0 = fmaxf(acc0[i] + b1, 0.0f);
                float r1 = fmaxf(acc1[i] + b1, 0.0f);
                d0[i] += r0 * w2;
                d1[i] += r1 * w2;
            }
        }

        // Reduce over N (lanes within each 16-lane half)
        #pragma unroll
        for (int i = 0; i < 8; ++i) {
            #pragma unroll
            for (int m = 1; m < 16; m <<= 1) {
                sA0[i] += __shfl_xor(sA0[i], m, 32);
                sB0[i] += __shfl_xor(sB0[i], m, 32);
                sA1[i] += __shfl_xor(sA1[i], m, 32);
                sB1r[i] += __shfl_xor(sB1r[i], m, 32);
            }
        }

        // Scatter per-pair MLP scalars via per-wave LDS scratch (32 pairs * 2)
        float* myMLP = sMLP + w * 64;
        if (lane == 0) {
            #pragma unroll
            for (int i = 0; i < 8; ++i) {
                myMLP[i * 2]            = sA0[i] + bA2;  myMLP[i * 2 + 1]        = sB0[i] + bB2;
                myMLP[(16 + i) * 2]     = sA1[i] + bA2;  myMLP[(16 + i) * 2 + 1] = sB1r[i] + bB2;
            }
        }
        if (lane == 16) {
            #pragma unroll
            for (int i = 0; i < 8; ++i) {
                myMLP[(8 + i) * 2]      = sA0[i] + bA2;  myMLP[(8 + i) * 2 + 1]  = sB0[i] + bB2;
                myMLP[(24 + i) * 2]     = sA1[i] + bA2;  myMLP[(24 + i) * 2 + 1] = sB1r[i] + bB2;
            }
        }

        // -------- physics per pair (lanes 0..15 own pairs; upper half masked) --------
        float vdwS = 0.0f, hbS = 0.0f, mtS = 0.0f, hpS = 0.0f;
        const float mask = (lane < 16) ? 1.0f : 0.0f;
        #pragma unroll
        for (int sub = 0; sub < 2; ++sub) {
            const int pl = sub * 16 + row;
            float mlpA = myMLP[pl * 2];                // same-wave LDS RAW: HW in-order + waitcnt
            float mlpB = myMLP[pl * 2 + 1];

            const int t = t0 + sub * 16 + row;
            const float tx = tpos[(b * NT + t) * 3 + 0];
            const float ty = tpos[(b * NT + t) * 3 + 1];
            const float tz = tpos[(b * NT + t) * 3 + 2];
            float dx = lx - tx, dy = ly - ty, dz = lz - tz;
            float dm = sqrtf(dx * dx + dy * dy + dz * dz + 1e-10f);
            dm = (dm < 0.5f) ? 1e10f : dm;

            float Bp   = tanhf(mlpB) * 0.2f;
            float dm0  = rl + trad[b * NT + t] + Bp;
            float dm0c = (dm0 < 1e-4f) ? 1.0f : dm0;
            float ratio = dm0c / dm;
            float r2 = ratio * ratio;
            float r6 = r2 * r2 * r2;
            float vdwE = fminf(r6 * r6 - 2.0f * r6, 100.0f) * nml * tnm[b * NT + t];
            float Av = 1.0f / (1.0f + __expf(-mlpA));
            Av = Av * (0.0356f - 0.0178f) + 0.0178f;

            float d = dm - dm0;
            const float* ib = inter + ((size_t)(b * 3) * NL + l) * NT + t;
            float I0 = ib[0];
            float I1 = ib[(size_t)NL * NT];
            float I2 = ib[2 * (size_t)NL * NT];

            vdwS += mask * (Av * vdwE);
            hbS  += mask * fminf(fmaxf(d * I0 * (-1.0f / 0.7f), 0.0f), 1.0f);
            mtS  += mask * fminf(fmaxf(d * I1 * (-1.0f / 0.7f), 0.0f), 1.0f);
            hpS  += mask * fminf(fmaxf((1.5f - d) * I2, 0.0f), 1.0f);
        }

        #pragma unroll
        for (int m = 1; m < 16; m <<= 1) {
            vdwS += __shfl_xor(vdwS, m, 32);
            hbS  += __shfl_xor(hbS,  m, 32);
            mtS  += __shfl_xor(mtS,  m, 32);
            hpS  += __shfl_xor(hpS,  m, 32);
        }
        if (lane == 0) {                               // lane0 holds sum of lanes 0..15
            atomicAdd(&sAcc[0], vdwS);
            atomicAdd(&sAcc[1], hbS);
            atomicAdd(&sAcc[2], mtS);
            atomicAdd(&sAcc[3], hpS);
        }
    }

    __syncthreads();
    if (tid < 4) {
        float* gAcc = (float*)(ws + WS_ACC);
        atomicAdd(&gAcc[b * 4 + tid], sAcc[tid]);
    }
}

// ---------------------------------------------------------------------------
// Finalize: apply coefficients and rotor scaling -> out[8][4]
// ---------------------------------------------------------------------------
__global__ void finalize_kernel(const unsigned char* __restrict__ ws,
                                const float* __restrict__ rotor,
                                const float* __restrict__ hbc,
                                const float* __restrict__ hpc,
                                const float* __restrict__ rc,
                                float* __restrict__ out) {
    int j = threadIdx.x;
    if (j >= NB * 4) return;
    int b = j >> 2, c = j & 3;
    const float* gAcc = (const float*)(ws + WS_ACC);
    float v = gAcc[j];
    float hb2 = hbc[0] * hbc[0];
    float hp2 = hpc[0] * hpc[0];
    float rc2 = rc[0] * rc[0];
    float e = (c == 0) ? v : ((c == 3) ? v * (-hp2) : v * (-hb2));
    out[j] = e / (1.0f + rc2 * rotor[b]);
}

// ---------------------------------------------------------------------------
extern "C" void kernel_launch(void* const* d_in, const int* in_sizes, int n_in,
                              void* d_out, int out_size, void* d_ws, size_t ws_size,
                              hipStream_t stream) {
    const float* ligand_pos   = (const float*)d_in[0];
    const float* target_pos   = (const float*)d_in[1];
    const float* ligand_rad   = (const float*)d_in[2];
    const float* target_rad   = (const float*)d_in[3];
    const float* ligand_nm    = (const float*)d_in[4];
    const float* target_nm    = (const float*)d_in[5];
    const float* inter        = (const float*)d_in[6];
    const float* rotor        = (const float*)d_in[7];
    const float* h_cat        = (const float*)d_in[8];
    const float* WA1          = (const float*)d_in[9];
    const float* bA1          = (const float*)d_in[10];
    const float* WA2          = (const float*)d_in[11];
    const float* bA2          = (const float*)d_in[12];
    const float* WB1          = (const float*)d_in[13];
    const float* bB1          = (const float*)d_in[14];
    const float* WB2          = (const float*)d_in[15];
    const float* bB2          = (const float*)d_in[16];
    const float* hbond_coeff  = (const float*)d_in[17];
    const float* hphob_coeff  = (const float*)d_in[18];
    const float* rotor_coeff  = (const float*)d_in[19];
    unsigned char* ws = (unsigned char*)d_ws;

    prep_kernel<<<256, 256, 0, stream>>>(WA1, bA1, WA2, WB1, bB1, WB2, ws);

    size_t smem = 131072 + 1024 + 1024 + 2048 + 16;   // W(f16) + b1 + w2 + mlp scratch + accum
    energy_kernel<<<NB * NL, 256, smem, stream>>>(h_cat, ligand_pos, target_pos,
                                                  ligand_rad, target_rad,
                                                  ligand_nm, target_nm, inter,
                                                  bA2, bB2, ws);

    finalize_kernel<<<1, 32, 0, stream>>>(ws, rotor, hbond_coeff, hphob_coeff,
                                          rotor_coeff, (float*)d_out);
}
